// SwinTransformer_8108898255021
// MI455X (gfx1250) — matrix operand order repaired and checked by reading, not compiled, not run on hardware
//
#include <hip/hip_runtime.h>
#include <math.h>

// ---------------- problem constants ----------------
#define BATCH   2
#define T_      16
#define H_      56
#define W_      56
#define C_      96
#define HEADS_  3
#define HD_     32
#define N_      392          // tokens per window (8*7*7)
#define NPAD    400          // padded to 25 tiles of 16
#define KPAD    416          // padded to 13 chunks of 32 (attention K dim)
#define NW_     128          // windows per batch (2*8*8)
#define BW      (BATCH*NW_)  // 256 windows total
#define HIDDEN_ 384
#define NTOK    (BATCH*T_*H_*W_)  // 100352 (= 6272 * 16)
#define SCALE_  0.17677669529663687f  // 32^-0.5

typedef __attribute__((ext_vector_type(16))) _Float16 v16h;
typedef __attribute__((ext_vector_type(8)))  _Float16 v8h;
typedef __attribute__((ext_vector_type(8)))  float    v8f;

// ---------------- WMMA fragment helpers (gfx1250 wave32 layouts) ----------------
// A (16x32 f16, MxK): lane l holds row m=l&15; halves e -> K = 16g+e,
// g = l>>4.  => two contiguous 16B chunks.
__device__ __forceinline__ v16h load_a_frag(const _Float16* A, int lda, int m0, int k0, int lane) {
  const _Float16* p = A + (size_t)(m0 + (lane & 15)) * lda + k0 + 16 * (lane >> 4);
  v8h lo = *(const v8h*)(p);
  v8h hi = *(const v8h*)(p + 8);
  v16h r;
#pragma unroll
  for (int i = 0; i < 8; ++i) { r[i] = lo[i]; r[8 + i] = hi[i]; }
  return r;
}

// B (32x16 f16, KxN): lane l holds column n=l&15; 16 contiguous K halves
// starting at K=16g.  BT is stored [N][K] row-major -> one 32B vector load.
__device__ __forceinline__ v16h load_b_frag(const _Float16* BT, int ldb, int n0, int k0, int lane) {
  const _Float16* p = BT + (size_t)(n0 + (lane & 15)) * ldb + k0 + 16 * (lane >> 4);
  return *(const v16h*)(p);
}

__device__ __forceinline__ v8f wmma_f16(v16h a, v16h b, v8f c) {
  return __builtin_amdgcn_wmma_f32_16x16x32_f16(false, a, false, b, (short)0, c, false, false);
}

// ---------------- small utility kernels ----------------
__global__ void k_f32to16(const float* __restrict__ s, _Float16* __restrict__ d, int n) {
  int i = blockIdx.x * blockDim.x + threadIdx.x;
  if (i < n) d[i] = (_Float16)s[i];
}

__global__ void k_zero16(_Float16* __restrict__ p, int n) {
  int i = blockIdx.x * blockDim.x + threadIdx.x;
  if (i < n) p[i] = (_Float16)0.f;
}

// ---------------- LN1 + roll(-4,-3,-3) + window partition -> f16 ----------------
// one wave per destination token; lane handles channels {lane, lane+32, lane+64}
__global__ __launch_bounds__(128) void k_ln1_partition(
    const float* __restrict__ x, const float* __restrict__ g, const float* __restrict__ b,
    _Float16* __restrict__ hwin) {
  int lane = threadIdx.x & 31;
  int d = blockIdx.x * 4 + (threadIdx.x >> 5);
  if (d >= BW * N_) return;
  int b_ = d / N_, nl = d % N_;
  int bb = b_ / NW_, widx = b_ % NW_;
  int tw = widx >> 6, hw = (widx >> 3) & 7, ww = widx & 7;
  int lt = nl / 49, lh = (nl / 7) % 7, lw = nl % 7;
  int t = (tw * 8 + lt + 4) & 15;
  int h = (hw * 7 + lh + 3) % 56;
  int w = (ww * 7 + lw + 3) % 56;
  const float* xp = x + (size_t)(((bb * T_ + t) * H_ + h) * W_ + w) * C_;
  float v0 = xp[lane], v1 = xp[lane + 32], v2 = xp[lane + 64];
  float s = v0 + v1 + v2, ss = v0 * v0 + v1 * v1 + v2 * v2;
#pragma unroll
  for (int o = 16; o >= 1; o >>= 1) { s += __shfl_xor(s, o); ss += __shfl_xor(ss, o); }
  float mean = s * (1.f / 96.f);
  float var  = ss * (1.f / 96.f) - mean * mean;
  float r = rsqrtf(var + 1e-5f);
  _Float16* hp = hwin + (size_t)(b_ * NPAD + nl) * C_;
  hp[lane]      = (_Float16)((v0 - mean) * r * g[lane]      + b[lane]);
  hp[lane + 32] = (_Float16)((v1 - mean) * r * g[lane + 32] + b[lane + 32]);
  hp[lane + 64] = (_Float16)((v2 - mean) * r * g[lane + 64] + b[lane + 64]);
}

// ---------------- QKV GEMM: (392x96) @ (96x288)^T per window ----------------
// one wave per 16x16 output tile.  Destination selection (q/k/v, head) is
// BLOCK-UNIFORM: a 16-wide tile never straddles a 96- or 32-channel boundary,
// so it is derived from nt (scalar) -> no per-lane divergence in the epilogue.
__global__ __launch_bounds__(32) void k_qkv(
    const _Float16* __restrict__ hwin, const _Float16* __restrict__ wq,
    const float* __restrict__ bias, _Float16* __restrict__ qb,
    _Float16* __restrict__ kb, _Float16* __restrict__ vT) {
  int lane = threadIdx.x;
  int nt = blockIdx.x, mt = blockIdx.y, b_ = blockIdx.z;
  const _Float16* A = hwin + (size_t)b_ * NPAD * C_;
  v8f acc = {};
#pragma unroll
  for (int kc = 0; kc < 3; ++kc) {
    if (kc < 2)  // prefetch next weight K-chunk (global_prefetch_b8)
      __builtin_prefetch(wq + (size_t)(nt * 16 + (lane & 15)) * C_ + (kc + 1) * 32, 0, 3);
    v16h a  = load_a_frag(A, C_, mt * 16, kc * 32, lane);
    v16h bf = load_b_frag(wq, C_, nt * 16, kc * 32, lane);
    acc = wmma_f16(a, bf, acc);
  }
  int which = nt / 6;            // 0=q 1=k 2=v   (uniform)
  int rem   = nt % 6;
  int head  = rem >> 1;          // uniform
  int hd    = (rem & 1) * 16 + (lane & 15);
  float bo  = bias[nt * 16 + (lane & 15)];
  if (which == 0) {
#pragma unroll
    for (int j = 0; j < 8; ++j) {
      int m = mt * 16 + j + 8 * (lane >> 4);
      qb[((size_t)(b_ * 3 + head) * NPAD + m) * HD_ + hd] = (_Float16)((acc[j] + bo) * SCALE_);
    }
  } else if (which == 1) {
#pragma unroll
    for (int j = 0; j < 8; ++j) {
      int m = mt * 16 + j + 8 * (lane >> 4);
      kb[((size_t)(b_ * 3 + head) * NPAD + m) * HD_ + hd] = (_Float16)(acc[j] + bo);
    }
  } else {
    _Float16* vp = vT + ((size_t)(b_ * 3 + head) * HD_ + hd) * KPAD;
#pragma unroll
    for (int j = 0; j < 8; ++j) {
      int m = mt * 16 + j + 8 * (lane >> 4);
      vp[m] = (_Float16)(acc[j] + bo);
    }
  }
}

// ---------------- attention: per (window, head, 16-row tile) ----------------
// S = q@k^T (+rel bias +region mask) into LDS, softmax, P@V, store O f16
__global__ __launch_bounds__(64) void k_attn(
    const _Float16* __restrict__ qbuf, const _Float16* __restrict__ kbuf,
    const _Float16* __restrict__ vT, const float* __restrict__ relb,
    _Float16* __restrict__ obuf) {
  int mt = blockIdx.x, head = blockIdx.y, b_ = blockIdx.z;
  int tid = threadIdx.x, lane = tid & 31, wv = tid >> 5;
  __shared__ __align__(32) float    S[16][NPAD];
  __shared__ __align__(32) _Float16 P[16][KPAD];
  __shared__ float red[16][4];
  __shared__ float rowmax[16], rowsum[16];
  __shared__ int   rowTHW[16][4];   // (t,h,w,region) per query row of this tile

  int widx = b_ % NW_;
  int tw = widx >> 6, hw = (widx >> 3) & 7, ww = widx & 7;
  const _Float16* q  = qbuf + (size_t)(b_ * 3 + head) * NPAD * HD_;
  const _Float16* k  = kbuf + (size_t)(b_ * 3 + head) * NPAD * HD_;
  const _Float16* vt = vT   + (size_t)(b_ * 3 + head) * HD_ * KPAD;

  // precompute query-row coordinate decompositions once per block
  if (tid < 16) {
    int ig = mt * 16 + tid;
    int ii = ig < N_ ? ig : N_ - 1;
    int ti = ii / 49, hi = (ii / 7) % 7, wi = ii % 7;
    int tgi = tw * 8 + ti, hgi = hw * 7 + hi, wgi = ww * 7 + wi;
    rowTHW[tid][0] = ti; rowTHW[tid][1] = hi; rowTHW[tid][2] = wi;
    rowTHW[tid][3] = (tgi < 8 ? 0 : (tgi < 12 ? 1 : 2)) * 9 +
                     (hgi < 49 ? 0 : (hgi < 53 ? 1 : 2)) * 3 +
                     (wgi < 49 ? 0 : (wgi < 53 ? 1 : 2));
  }
  __syncthreads();

  // ---- phase 1: score tiles (K = HD = 32 -> single WMMA each) ----
  v16h afrag = load_a_frag(q, HD_, mt * 16, 0, lane);
  for (int ntile = wv; ntile < 25; ntile += 2) {
    v16h bf = load_b_frag(k, HD_, ntile * 16, 0, lane);
    v8f acc = {};
    acc = wmma_f16(afrag, bf, acc);
    int nn = ntile * 16 + (lane & 15);
    int jj = nn < N_ ? nn : N_ - 1;
    int tj = jj / 49, hj = (jj / 7) % 7, wj = jj % 7;
    int tgj = tw * 8 + tj, hgj = hw * 7 + hj, wgj = ww * 7 + wj;
    int rj = (tgj < 8 ? 0 : (tgj < 12 ? 1 : 2)) * 9 +
             (hgj < 49 ? 0 : (hgj < 53 ? 1 : 2)) * 3 +
             (wgj < 49 ? 0 : (wgj < 53 ? 1 : 2));
#pragma unroll
    for (int j = 0; j < 8; ++j) {
      int m = j + 8 * (lane >> 4);
      // faithful to reference index math: idx = dt*15 + dh*13 + dw
      int idx = (rowTHW[m][0] - tj + 7) * 15 +
                (rowTHW[m][1] - hj + 6) * 13 +
                (rowTHW[m][2] - wj + 6);
      float bm = relb[idx * HEADS_ + head] + (rowTHW[m][3] == rj ? 0.f : -100.f);
      S[m][nn] = acc[j] + bm;
    }
  }
  __syncthreads();

  // ---- phase 2: softmax over cols [0,392), P padded with zeros ----
  int r = tid >> 2, c4 = tid & 3;
  float mx = -1e30f;
  for (int c = c4; c < N_; c += 4) mx = fmaxf(mx, S[r][c]);
  red[r][c4] = mx;
  __syncthreads();
  if (tid < 16) {
    float m = fmaxf(fmaxf(red[tid][0], red[tid][1]), fmaxf(red[tid][2], red[tid][3]));
    rowmax[tid] = m;
  }
  __syncthreads();
  float rm = rowmax[r];
  float sm = 0.f;
  for (int c = c4; c < N_; c += 4) {
    float e = __expf(S[r][c] - rm);
    P[r][c] = (_Float16)e;
    sm += e;
  }
  for (int c = N_ + c4; c < KPAD; c += 4) P[r][c] = (_Float16)0.f;
  red[r][c4] = sm;
  __syncthreads();
  if (tid < 16) rowsum[tid] = red[tid][0] + red[tid][1] + red[tid][2] + red[tid][3];
  __syncthreads();

  // ---- phase 3: O = P @ V, 1/rowsum folded into epilogue ----
  if (wv < 2) {
    v8f acc = {};
    for (int kc = 0; kc < 13; ++kc) {
      if (kc < 12)
        __builtin_prefetch(vt + (size_t)(wv * 16 + (lane & 15)) * KPAD + (kc + 1) * 32, 0, 3);
      v16h a  = load_a_frag(&P[0][0], KPAD, 0, kc * 32, lane);
      v16h bf = load_b_frag(vt, KPAD, wv * 16, kc * 32, lane);
      acc = wmma_f16(a, bf, acc);
    }
    int n = wv * 16 + (lane & 15);
#pragma unroll
    for (int j = 0; j < 8; ++j) {
      int m = j + 8 * (lane >> 4);
      int tok = mt * 16 + m;
      float v = acc[j] / rowsum[m];
      obuf[((size_t)b_ * NPAD + tok) * C_ + head * HD_ + n] = (_Float16)v;
    }
  }
}

// ---------------- proj GEMM + window reverse + roll back + residual ----------------
__global__ __launch_bounds__(32) void k_proj(
    const _Float16* __restrict__ obuf, const _Float16* __restrict__ wp,
    const float* __restrict__ pb, const float* __restrict__ x, float* __restrict__ out) {
  int lane = threadIdx.x;
  int nt = blockIdx.x, mt = blockIdx.y, b_ = blockIdx.z;
  const _Float16* A = obuf + (size_t)b_ * NPAD * C_;
  v8f acc = {};
#pragma unroll
  for (int kc = 0; kc < 3; ++kc) {
    v16h a  = load_a_frag(A, C_, mt * 16, kc * 32, lane);
    v16h bf = load_b_frag(wp, C_, nt * 16, kc * 32, lane);
    acc = wmma_f16(a, bf, acc);
  }
  int o = nt * 16 + (lane & 15);
  int bb = b_ / NW_, widx = b_ % NW_;
  int tw = widx >> 6, hw = (widx >> 3) & 7, ww = widx & 7;
  float bo = pb[o];
#pragma unroll
  for (int j = 0; j < 8; ++j) {
    int m = mt * 16 + j + 8 * (lane >> 4);
    if (m < N_) {
      int lt = m / 49, lh = (m / 7) % 7, lw = m % 7;
      int t = (tw * 8 + lt + 4) & 15;
      int h = (hw * 7 + lh + 3) % 56;
      int w = (ww * 7 + lw + 3) % 56;
      size_t idx = (size_t)(((bb * T_ + t) * H_ + h) * W_ + w) * C_ + o;
      out[idx] = x[idx] + acc[j] + bo;
    }
  }
}

// ---------------- fused MLP: LN2 + fc1 + GELU + fc2 + residual ----------------
// one block of 8 waves per 16-token row tile; hidden tensor lives in LDS only
__global__ __launch_bounds__(256) void k_mlp(
    float* __restrict__ xo, const float* __restrict__ g, const float* __restrict__ b,
    const _Float16* __restrict__ w1, const float* __restrict__ b1,
    const _Float16* __restrict__ w2, const float* __restrict__ b2) {
  __shared__ __align__(32) _Float16 h16[16][C_];
  __shared__ __align__(32) _Float16 g16[16][HIDDEN_];
  __shared__ float redS[16][16], redQ[16][16];
  __shared__ float mean_[16], rstd_[16];
  int tid = threadIdx.x, lane = tid & 31, wv = tid >> 5;
  int row0 = blockIdx.x * 16;

  // LN2 (16 threads per token)
  int r = tid >> 4, c16 = tid & 15;
  const float* xr = xo + (size_t)(row0 + r) * C_;
  float s = 0.f, ss = 0.f;
  for (int c = c16; c < C_; c += 16) { float v = xr[c]; s += v; ss += v * v; }
  redS[r][c16] = s; redQ[r][c16] = ss;
  __syncthreads();
  if (tid < 16) {
    float sm = 0.f, sq = 0.f;
#pragma unroll
    for (int i = 0; i < 16; ++i) { sm += redS[tid][i]; sq += redQ[tid][i]; }
    float mn = sm * (1.f / 96.f);
    mean_[tid] = mn;
    rstd_[tid] = rsqrtf(sq * (1.f / 96.f) - mn * mn + 1e-5f);
  }
  __syncthreads();
  float mn = mean_[r], rs = rstd_[r];
  for (int c = c16; c < C_; c += 16)
    h16[r][c] = (_Float16)((xr[c] - mn) * rs * g[c] + b[c]);
  __syncthreads();

  // fc1 (16x96 @ 96x384) + exact GELU -> LDS f16
  for (int nt = wv; nt < 24; nt += 8) {
    v8f acc = {};
#pragma unroll
    for (int kc = 0; kc < 3; ++kc) {
      if (kc < 2)
        __builtin_prefetch(w1 + (size_t)(nt * 16 + (lane & 15)) * C_ + (kc + 1) * 32, 0, 3);
      v16h a  = load_a_frag(&h16[0][0], C_, 0, kc * 32, lane);
      v16h bf = load_b_frag(w1, C_, nt * 16, kc * 32, lane);
      acc = wmma_f16(a, bf, acc);
    }
    int o = nt * 16 + (lane & 15);
    float bo = b1[o];
#pragma unroll
    for (int j = 0; j < 8; ++j) {
      int m = j + 8 * (lane >> 4);
      float v = acc[j] + bo;
      v = 0.5f * v * (1.f + erff(v * 0.70710678118654752f));
      g16[m][o] = (_Float16)v;
    }
  }
  __syncthreads();

  // fc2 (16x384 @ 384x96) + residual into d_out
  if (wv < 6) {
    v8f acc = {};
#pragma unroll
    for (int kc = 0; kc < 12; ++kc) {
      if (kc < 11)
        __builtin_prefetch(w2 + (size_t)(wv * 16 + (lane & 15)) * HIDDEN_ + (kc + 1) * 32, 0, 3);
      v16h a  = load_a_frag(&g16[0][0], HIDDEN_, 0, kc * 32, lane);
      v16h bf = load_b_frag(w2, HIDDEN_, wv * 16, kc * 32, lane);
      acc = wmma_f16(a, bf, acc);
    }
    int o = wv * 16 + (lane & 15);
    float bo = b2[o];
#pragma unroll
    for (int j = 0; j < 8; ++j) {
      int m = j + 8 * (lane >> 4);
      size_t idx = (size_t)(row0 + m) * C_ + o;
      xo[idx] = xo[idx] + acc[j] + bo;
    }
  }
}

// ---------------- host-side orchestration ----------------
extern "C" void kernel_launch(void* const* d_in, const int* in_sizes, int n_in,
                              void* d_out, int out_size, void* d_ws, size_t ws_size,
                              hipStream_t stream) {
  (void)in_sizes; (void)n_in; (void)out_size; (void)ws_size;
  const float* x      = (const float*)d_in[0];
  const float* n1g    = (const float*)d_in[1];
  const float* n1b    = (const float*)d_in[2];
  const float* qkv_w  = (const float*)d_in[3];
  const float* qkv_b  = (const float*)d_in[4];
  const float* rel_b  = (const float*)d_in[5];
  const float* proj_w = (const float*)d_in[6];
  const float* proj_b = (const float*)d_in[7];
  const float* n2g    = (const float*)d_in[8];
  const float* n2b    = (const float*)d_in[9];
  const float* fc1_w  = (const float*)d_in[10];
  const float* fc1_b  = (const float*)d_in[11];
  const float* fc2_w  = (const float*)d_in[12];
  const float* fc2_b  = (const float*)d_in[13];
  float* out = (float*)d_out;

  char* base = (char*)d_ws;
  size_t off = 0;
  auto alloc = [&](size_t bytes) -> char* {
    char* p = base + off;
    off += (bytes + 255) & ~(size_t)255;
    return p;
  };
  _Float16* wq16 = (_Float16*)alloc((size_t)288 * 96 * 2);
  _Float16* wp16 = (_Float16*)alloc((size_t)96 * 96 * 2);
  _Float16* w116 = (_Float16*)alloc((size_t)384 * 96 * 2);
  _Float16* w216 = (_Float16*)alloc((size_t)96 * 384 * 2);
  _Float16* hwin = (_Float16*)alloc((size_t)BW * NPAD * C_ * 2);       // aliased as obuf
  _Float16* qb   = (_Float16*)alloc((size_t)BW * 3 * NPAD * HD_ * 2);
  _Float16* kb   = (_Float16*)alloc((size_t)BW * 3 * NPAD * HD_ * 2);
  _Float16* vT   = (_Float16*)alloc((size_t)BW * 3 * HD_ * KPAD * 2);
  _Float16* obuf = hwin;  // hwin is fully consumed by k_qkv before k_attn writes obuf

  k_f32to16<<<(288 * 96 + 255) / 256, 256, 0, stream>>>(qkv_w, wq16, 288 * 96);
  k_f32to16<<<(96 * 96 + 255) / 256, 256, 0, stream>>>(proj_w, wp16, 96 * 96);
  k_f32to16<<<(384 * 96 + 255) / 256, 256, 0, stream>>>(fc1_w, w116, 384 * 96);
  k_f32to16<<<(96 * 384 + 255) / 256, 256, 0, stream>>>(fc2_w, w216, 96 * 384);

  int nhw = BW * NPAD * C_;
  k_zero16<<<(nhw + 255) / 256, 256, 0, stream>>>(hwin, nhw);
  int nvt = BW * 3 * HD_ * KPAD;
  k_zero16<<<(nvt + 255) / 256, 256, 0, stream>>>(vT, nvt);

  k_ln1_partition<<<(BW * N_) / 4, 128, 0, stream>>>(x, n1g, n1b, hwin);
  k_qkv<<<dim3(18, 25, BW), 32, 0, stream>>>(hwin, wq16, qkv_b, qb, kb, vT);
  k_attn<<<dim3(25, 3, BW), 64, 0, stream>>>(qb, kb, vT, rel_b, obuf);
  k_proj<<<dim3(6, 25, BW), 32, 0, stream>>>(obuf, wp16, proj_b, x, out);
  k_mlp<<<NTOK / 16, 256, 0, stream>>>(out, n2g, n2b, w116, fc1_b, w216, fc2_b);
}
